// GraphAttentionLayer_64407329570939
// MI455X (gfx1250) — matrix operand order repaired and checked by reading, not compiled, not run on hardware
//
#include <hip/hip_runtime.h>

typedef __attribute__((ext_vector_type(16))) _Float16 v16h;
typedef __attribute__((ext_vector_type(8)))  float    v8f;

#define GAT_B    8
#define GAT_N    2048
#define GAT_F    256
#define GAT_ROWS (GAT_B * GAT_N)   // 16384
#define LRELU_ALPHA 0.2f

// ---------------------------------------------------------------------------
// Kernel A: convert W (Fout x Fin, f32 row-major) -> f16 row-major.
// B-tile for projection WMMA reads W16[o][k0+lb8 .. +7] and W16[o][k0+lb8+16 .. +7].
// ---------------------------------------------------------------------------
__global__ void gat_wcvt(const float* __restrict__ W, _Float16* __restrict__ W16, int n) {
  int i = blockIdx.x * blockDim.x + threadIdx.x;
  if (i < n) W16[i] = (_Float16)W[i];
}

// ---------------------------------------------------------------------------
// Kernel B: projection. One wave per 16-row tile of the flattened (B*N) rows.
//   Wh = h @ W^T   (f32 accum via v_wmma_f32_16x16x32_f16)
// Fused outputs:
//   WhT  : f16, [F][B*N] feature-major (B-operand-ready for the attn GEMM)
//   fvec : Wh @ a1   (f32, per row)
//   gvec : Wh @ a2   (f32, per row)
// ---------------------------------------------------------------------------
__global__ __launch_bounds__(256)
void gat_proj(const float* __restrict__ h,          // [B*N][F]
              const _Float16* __restrict__ W16,     // [F][F]
              const float* __restrict__ a,          // [2F]
              _Float16* __restrict__ WhT,           // [F][B*N]
              float* __restrict__ fvec,             // [B*N]
              float* __restrict__ gvec)             // [B*N]
{
  const int lane = threadIdx.x & 31;
  const int wave = threadIdx.x >> 5;
  const int tile = blockIdx.x * 8 + wave;           // 0..1023
  const int g0   = tile * 16;                       // base global row
  const int m    = lane & 15;
  const int hi   = lane >> 4;
  const int lb8  = hi * 8;                          // A-layout K base per half
  const int kb16 = hi * 16;                         // unused (B also uses lb8)

  v8f acc[16] = {};                                 // 16 o-tiles x 8 f32

  const float* hrow = h + (size_t)(g0 + m) * GAT_F;

  for (int k0 = 0; k0 < GAT_F; k0 += 32) {
    // A tile: rows of h, f32 -> f16. Two runs of 8 consecutive K per lane.
    const float4 x0 = *(const float4*)(hrow + k0 + lb8);
    const float4 x1 = *(const float4*)(hrow + k0 + lb8 + 4);
    const float4 x2 = *(const float4*)(hrow + k0 + lb8 + 16);
    const float4 x3 = *(const float4*)(hrow + k0 + lb8 + 20);
    v16h pa;
    pa[0]  = (_Float16)x0.x; pa[1]  = (_Float16)x0.y;
    pa[2]  = (_Float16)x0.z; pa[3]  = (_Float16)x0.w;
    pa[4]  = (_Float16)x1.x; pa[5]  = (_Float16)x1.y;
    pa[6]  = (_Float16)x1.z; pa[7]  = (_Float16)x1.w;
    pa[8]  = (_Float16)x2.x; pa[9]  = (_Float16)x2.y;
    pa[10] = (_Float16)x2.z; pa[11] = (_Float16)x2.w;
    pa[12] = (_Float16)x3.x; pa[13] = (_Float16)x3.y;
    pa[14] = (_Float16)x3.z; pa[15] = (_Float16)x3.w;

#pragma unroll
    for (int t = 0; t < 16; ++t) {
      // B(k, n=o) = W[o][k]: two runs of 8 consecutive f16 per lane (K = lb8.., lb8+16..).
      const _Float16* wp = W16 + (size_t)(t * 16 + m) * GAT_F + k0 + lb8;
      const v16h bt = __builtin_shufflevector(*(const __attribute__((ext_vector_type(8))) _Float16*)(wp), *(const __attribute__((ext_vector_type(8))) _Float16*)(wp + 16), 0, 1, 2, 3, 4, 5, 6, 7, 8, 9, 10, 11, 12, 13, 14, 15);
      acc[t] = __builtin_amdgcn_wmma_f32_16x16x32_f16(
          false, pa, false, bt, (short)0, acc[t], false, false);
    }
  }

  // Fused f = Wh@a1, g = Wh@a2. C layout: acc[t][r] = Wh[row g0+lb8+r][t*16+m].
  float pf[8] = {}, pg[8] = {};
#pragma unroll
  for (int t = 0; t < 16; ++t) {
    const float a1 = a[t * 16 + m];
    const float a2 = a[GAT_F + t * 16 + m];
#pragma unroll
    for (int r = 0; r < 8; ++r) {
      pf[r] += acc[t][r] * a1;
      pg[r] += acc[t][r] * a2;
    }
  }
#pragma unroll
  for (int mask = 1; mask < 16; mask <<= 1) {
#pragma unroll
    for (int r = 0; r < 8; ++r) {
      pf[r] += __shfl_xor(pf[r], mask, 32);
      pg[r] += __shfl_xor(pg[r], mask, 32);
    }
  }
  if (m == 0) {
#pragma unroll
    for (int r = 0; r < 8; ++r) {
      fvec[g0 + lb8 + r] = pf[r];
      gvec[g0 + lb8 + r] = pg[r];
    }
  }

  // WhT f16 store: lane's 8 accum values for a fixed feature column are
  // consecutive node indices -> one 16-byte store per o-tile.
#pragma unroll
  for (int t = 0; t < 16; ++t) {
    union { _Float16 hh[8]; uint4 u; } pk;
#pragma unroll
    for (int r = 0; r < 8; ++r) pk.hh[r] = (_Float16)acc[t][r];
    *(uint4*)(WhT + (size_t)(t * 16 + m) * GAT_ROWS + g0 + lb8) = pk.u;
  }
}

// ---------------------------------------------------------------------------
// Kernel C: masked-softmax attention + aggregation, single pass over j.
// One wave per 16-row tile: acc += P_chunk(16x32 f16) @ WhT_chunk(32x16 f16)
// with P = exp(leakyrelu(f_i + g_j)) * (adj != 0), unnormalized; divide by the
// row sum at the end (e is bounded, so no max subtraction needed in fp32).
// ---------------------------------------------------------------------------
__global__ __launch_bounds__(256)
void gat_attn(const int* __restrict__ adj,          // [B][N][N]
              const _Float16* __restrict__ WhT,     // [F][B*N]
              const float* __restrict__ fvec,       // [B*N]
              const float* __restrict__ gvec,       // [B*N]
              float* __restrict__ out)              // [B*N][F]
{
  const int lane = threadIdx.x & 31;
  const int wave = threadIdx.x >> 5;
  const int tile = blockIdx.x * 8 + wave;           // 0..1023
  const int b    = tile >> 7;                       // 128 tiles per batch
  const int i0   = (tile & 127) << 4;
  const int m    = lane & 15;
  const int hi   = lane >> 4;
  const int lb8  = hi * 8;
  const int kb16 = hi * 16;

  const float fi = fvec[(size_t)b * GAT_N + i0 + m];
  const float* gb = gvec + (size_t)b * GAT_N;
  const int* adjRow = adj + ((size_t)b * GAT_N + i0 + m) * GAT_N;
  const _Float16* whb = WhT + (size_t)b * GAT_N;

  v8f acc[16] = {};
  float s = 0.f;

  for (int j0 = 0; j0 < GAT_N; j0 += 32) {
    const int4   q0 = *(const int4*)(adjRow + j0 + lb8);
    const int4   q1 = *(const int4*)(adjRow + j0 + lb8 + 4);
    const int4   q2 = *(const int4*)(adjRow + j0 + lb8 + 16);
    const int4   q3 = *(const int4*)(adjRow + j0 + lb8 + 20);
    const float4 y0 = *(const float4*)(gb + j0 + lb8);
    const float4 y1 = *(const float4*)(gb + j0 + lb8 + 4);
    const float4 y2 = *(const float4*)(gb + j0 + lb8 + 16);
    const float4 y3 = *(const float4*)(gb + j0 + lb8 + 20);

    if (j0 + 32 < GAT_N) __builtin_prefetch(adjRow + j0 + 32 + lb8, 0, 1);

    v16h pa;
    const int   av[16] = { q0.x, q0.y, q0.z, q0.w, q1.x, q1.y, q1.z, q1.w,
                           q2.x, q2.y, q2.z, q2.w, q3.x, q3.y, q3.z, q3.w };
    const float gv[16] = { y0.x, y0.y, y0.z, y0.w, y1.x, y1.y, y1.z, y1.w,
                           y2.x, y2.y, y2.z, y2.w, y3.x, y3.y, y3.z, y3.w };
#pragma unroll
    for (int e = 0; e < 16; ++e) {
      float ev = fi + gv[e];
      ev = ev > 0.f ? ev : LRELU_ALPHA * ev;
      const float p = (av[e] != 0) ? __expf(ev) : 0.f;
      s += p;
      pa[e] = (_Float16)p;
    }

#pragma unroll
    for (int t = 0; t < 16; ++t) {
      const _Float16* wp = whb + (size_t)(t * 16 + m) * GAT_ROWS + j0 + lb8;
      const v16h bt = __builtin_shufflevector(*(const __attribute__((ext_vector_type(8))) _Float16*)(wp), *(const __attribute__((ext_vector_type(8))) _Float16*)(wp + 16), 0, 1, 2, 3, 4, 5, 6, 7, 8, 9, 10, 11, 12, 13, 14, 15);
      acc[t] = __builtin_amdgcn_wmma_f32_16x16x32_f16(
          false, pa, false, bt, (short)0, acc[t], false, false);
    }
  }

  // Row sums: lanes L and L+16 cover complementary K halves of row (L&15).
  const float stot = s + __shfl_xor(s, 16, 32);
  float rs[8];
#pragma unroll
  for (int r = 0; r < 8; ++r) {
    const float v = __shfl(stot, lb8 + r, 32);   // row sum for row lb8 + r
    rs[r] = v > 0.f ? 1.0f / v : 0.f;
  }

  float* orow = out + ((size_t)b * GAT_N + i0) * GAT_F;
#pragma unroll
  for (int t = 0; t < 16; ++t) {
#pragma unroll
    for (int r = 0; r < 8; ++r) {
      orow[(size_t)(lb8 + r) * GAT_F + t * 16 + m] = acc[t][r] * rs[r];
    }
  }
}

// ---------------------------------------------------------------------------
extern "C" void kernel_launch(void* const* d_in, const int* in_sizes, int n_in,
                              void* d_out, int out_size, void* d_ws, size_t ws_size,
                              hipStream_t stream) {
  (void)in_sizes; (void)n_in; (void)out_size; (void)ws_size;
  const float* h   = (const float*)d_in[0];   // [8][2048][256]
  const int*   adj = (const int*)d_in[1];     // [8][2048][2048]
  const float* W   = (const float*)d_in[2];   // [256][256]
  const float* a   = (const float*)d_in[3];   // [512]
  float* out = (float*)d_out;                 // [8][2048][256]

  // Workspace layout
  char* ws = (char*)d_ws;
  _Float16* W16  = (_Float16*)ws;                                  // 128 KB
  _Float16* WhT  = (_Float16*)(ws + (size_t)GAT_F * GAT_F * 2);    // 8 MB
  float*    fvec = (float*)(ws + (size_t)GAT_F * GAT_F * 2
                               + (size_t)GAT_F * GAT_ROWS * 2);    // 64 KB
  float*    gvec = fvec + GAT_ROWS;                                // 64 KB

  gat_wcvt<<<(GAT_F * GAT_F + 255) / 256, 256, 0, stream>>>(W, W16, GAT_F * GAT_F);
  gat_proj<<<GAT_ROWS / 16 / 8, 256, 0, stream>>>(h, W16, a, WhT, fvec, gvec);
  gat_attn<<<GAT_ROWS / 16 / 8, 256, 0, stream>>>(adj, WhT, fvec, gvec, out);
}
